// TemporalFeatureAggregator_30133490548787
// MI455X (gfx1250) — matrix operand order repaired and checked by reading, not compiled, not run on hardware
//
#include <hip/hip_runtime.h>
#include <hip/hip_bf16.h>
#include <stdint.h>

typedef __attribute__((ext_vector_type(16))) _Float16 v16h;
typedef __attribute__((ext_vector_type(8)))  _Float16 v8h;
typedef __attribute__((ext_vector_type(8)))  float    v8f;

#define NEG_SLOPE 0.01f
#define XT 264   // xs_t row stride (halves): 68 rows x 264, 528B rows -> 16B aligned, bank-spread
#define HT 72    // hdn row stride (halves): 66 rows x 72, 144B rows -> 16B aligned, bank-spread

// ---------------------------------------------------------------------------
// WMMA fragment loaders (wave32, v_wmma_f32_16x16x32_f16 layouts per CDNA5 ISA)
// ---------------------------------------------------------------------------

// A-matrix 16x32 f16 fragment from LDS; rows contiguous, K pairs -> ds_load_b32.
__device__ __forceinline__ v16h a_frag_lds(const _Float16* base, int stride, int lane) {
  union { v16h v; uint32_t u[8]; } f;
  const int m   = lane & 15;
  const int off = (lane & 16) ? 8 : 0;
  const uint32_t* row = (const uint32_t*)(base + m * stride);
#pragma unroll
  for (int j = 0; j < 8; ++j) {
    const int k = (j < 4) ? (off + 2 * j) : (16 + off + 2 * (j - 4));
    f.u[j] = row[k >> 1];
  }
  return f.v;
}

// B-matrix 32x16 f16 fragment, source rows indexed by s, K contiguous in row.
// Used for both xs_t (conv-a, row = s+dk+1) and hdn (conv-b, row = s+rowoff).
__device__ __forceinline__ v16h b_frag_rows(const _Float16* buf, int rstride, int k0,
                                            int row0, int lane) {
  union { v16h v; uint32_t u[8]; } f;
  const int row = row0 + (lane & 15);
  const int kb  = (lane & 16) ? 8 : 0;
  const uint32_t* src = (const uint32_t*)(buf + row * rstride + k0 + kb);
#pragma unroll
  for (int j = 0; j < 8; ++j) f.u[j] = src[(j < 4) ? j : (j + 4)];
  return f.v;
}

__device__ __forceinline__ uint32_t pack2h(float a, float b) {
  _Float16 h0 = (_Float16)a, h1 = (_Float16)b;
  union { _Float16 h; uint16_t u; } c0{h0}, c1{h1};
  return (uint32_t)c0.u | ((uint32_t)c1.u << 16);
}

#define WMMA_F16(A, B, C) \
  __builtin_amdgcn_wmma_f32_16x16x32_f16(false, (A), false, (B), (short)0, (C), false, false)

// ---------------------------------------------------------------------------
// Fully fused kernel: one block (8 wave32) per (p, n).
// ---------------------------------------------------------------------------
__global__ __launch_bounds__(256)
void tfa_fused_kernel(const float* __restrict__ x,
                      const float* __restrict__ w3x1_a,
                      const float* __restrict__ w3x1_b,
                      const float* __restrict__ w3x3_a,
                      const float* __restrict__ w3x3_b,
                      float* __restrict__ out) {
  constexpr int C = 256, H = 64;

  __shared__ __align__(16) _Float16 xs_t[68 * XT];    // x^T f16, row = s+2 (halo 2): 35.9 KB
  __shared__ __align__(16) _Float16 hdn[66 * HT];     // hdn^T f16, row = s+1 (halo 1): 9.5 KB
  __shared__ __align__(16) _Float16 wstage[256 * 32]; // weight staging: 16 KB

  const int tid  = threadIdx.x;
  const int lane = tid & 31;
  const int w    = tid >> 5;          // wave id 0..7
  const int pn   = blockIdx.x;        // p*128 + n
  const int p    = pn >> 7;

  const float* xrow = x + (size_t)pn * C * 64;

  // ---- stage x row -> LDS transposed f16 (zero halo), zero hdn halo rows ----
#pragma unroll 4
  for (int i = 0; i < 68; ++i) {
    const int s = i - 2;
    float v = 0.f;
    if (s >= 0 && s < 64) v = xrow[(size_t)tid * 64 + s];
    xs_t[i * XT + tid] = (_Float16)v;
  }
  if (tid < 64) { hdn[0 * HT + tid] = (_Float16)0.f; hdn[65 * HT + tid] = (_Float16)0.f; }

  const int mtA  = w & 3;             // conv-a: wave -> h-tile
  const int ntA0 = (w >> 2) * 2;      // conv-a: wave -> 2 s-tiles
  const int sA0  = ntA0 * 16 + (lane & 15);
  const int hiA  = (lane & 16) ? 8 : 0;

  v8f feat[8];                        // (2 c-tiles x 4 s-tiles) per wave

#pragma unroll
  for (int b = 0; b < 2; ++b) {
    const float* wab = (b == 0) ? (w3x1_a + (size_t)p * H * C * 3)
                                : (w3x3_a + (size_t)p * H * C * 3);
    const float* wbb = (b == 0) ? (w3x1_b + (size_t)p * C * H * 1)
                                : (w3x3_b + (size_t)p * C * H * 3);

    // =================== conv-a: hdn[64h x 64s] = wa x im2col(x) ===========
    v8f accA[2] = {};
    for (int kc = 0; kc < 12; ++kc) {           // K = 768, chunks of 64 (fixed dk)
      __syncthreads();                           // wstage reuse fence
      {
        const int dk   = kc >> 2;
        const int cin0 = (kc & 3) * 64;
#pragma unroll
        for (int i = 0; i < 8; ++i) {            // stage wa chunk -> wstage[64][64], pk f16x2
          const int e2 = (i * 256 + tid) * 2;
          const int m  = e2 >> 6;
          const int kk = e2 & 63;
          const float f0 = wab[((size_t)m * C + (cin0 + kk))     * 3 + dk];
          const float f1 = wab[((size_t)m * C + (cin0 + kk + 1)) * 3 + dk];
          ((uint32_t*)wstage)[i * 256 + tid] = pack2h(f0, f1);
        }
        if (kc + 1 < 12) {                       // prefetch next chunk -> global_prefetch_b8
          const int dkn = (kc + 1) >> 2, cn0 = ((kc + 1) & 3) * 64;
          __builtin_prefetch(&wab[((size_t)(tid >> 2) * C + (cn0 + (tid & 3) * 16)) * 3 + dkn], 0, 1);
        }
      }
      __syncthreads();
      const int dk   = kc >> 2;
      const int cin0 = (kc & 3) * 64;
#pragma unroll
      for (int ks = 0; ks < 2; ++ks) {           // two k=32 steps per chunk
        const v16h a = a_frag_lds(&wstage[mtA * 16 * 64 + ks * 32], 64, lane);
        const int cinb = cin0 + ks * 32;
#pragma unroll
        for (int t = 0; t < 2; ++t) {            // B[k=cin][s] = xs_t[s+dk+1][cin]
          const v16h bf = b_frag_rows(xs_t, XT, cinb, (ntA0 + t) * 16 + dk + 1, lane);
          accA[t] = WMMA_F16(a, bf, accA[t]);
        }
      }
    }
    // LeakyReLU -> hdn^T (f16), one ds_store_b128 per tile
    __syncthreads();                             // prior readers of hdn done
#pragma unroll
    for (int t = 0; t < 2; ++t) {
      const int s     = sA0 + t * 16;
      const int hbase = mtA * 16 + hiA;
      v8h pk;
#pragma unroll
      for (int r = 0; r < 8; ++r) {
        float v = accA[t][r];
        v = (v >= 0.f) ? v : NEG_SLOPE * v;
        pk[r] = (_Float16)v;
      }
      *(v8h*)&hdn[(s + 1) * HT + hbase] = pk;
    }
    __syncthreads();

    // =================== conv-b: scores[256c x 64s] = wb x hdn =============
    v8f accS[8] = {};                            // [mt2(2)][nt(4)]
    const int nsteps = b ? 6 : 2;                // K = 192 / 64, k-steps of 32
    for (int st = 0; st < nsteps; ++st) {
      __syncthreads();
#pragma unroll
      for (int i = 0; i < 16; ++i) {             // stage wb slice -> wstage[256][32], pk f16x2
        const int e2 = (i * 256 + tid) * 2;
        const int cI = e2 >> 5;
        const int kk = e2 & 31;
        const int Kabs = st * 32 + kk;
        float f0, f1;
        if (b == 0) {
          f0 = wbb[(size_t)cI * H + Kabs];
          f1 = wbb[(size_t)cI * H + Kabs + 1];
        } else {
          const int dk2 = Kabs >> 6, hh = Kabs & 63;
          f0 = wbb[((size_t)cI * H + hh)     * 3 + dk2];
          f1 = wbb[((size_t)cI * H + hh + 1) * 3 + dk2];
        }
        ((uint32_t*)wstage)[i * 256 + tid] = pack2h(f0, f1);
      }
      __syncthreads();
      const int rowoff = b ? (st >> 1) : 1;      // hdn row = s + dk2 (halo folded in)
      const int hbaseB = b ? ((st & 1) * 32) : (st * 32);
      const v16h aA = a_frag_lds(&wstage[(2 * w + 0) * 16 * 32], 32, lane);
      const v16h aB = a_frag_lds(&wstage[(2 * w + 1) * 16 * 32], 32, lane);
#pragma unroll
      for (int nt = 0; nt < 4; ++nt) {
        const v16h bf = b_frag_rows(hdn, HT, hbaseB, nt * 16 + rowoff, lane);
        accS[nt]     = WMMA_F16(aA, bf, accS[nt]);
        accS[4 + nt] = WMMA_F16(aB, bf, accS[4 + nt]);
      }
    }

    // ============ epilogue: (avg+max pool)(x) * sigmoid(scores) ============
#pragma unroll
    for (int t = 0; t < 8; ++t) {
      const int mt2 = t >> 2, nt = t & 3;
      const int c0  = 32 * w + 16 * mt2 + ((lane & 16) ? 8 : 0);   // 8-aligned
      const int s   = nt * 16 + (lane & 15);
      // 5 aligned b128 tap loads cover r = 0..7 (c contiguous in xs_t rows)
      v8h tp0 = *(const v8h*)&xs_t[(s + 0) * XT + c0];
      v8h tp1 = *(const v8h*)&xs_t[(s + 1) * XT + c0];
      v8h tp2 = *(const v8h*)&xs_t[(s + 2) * XT + c0];
      v8h tp3 = *(const v8h*)&xs_t[(s + 3) * XT + c0];
      v8h tp4 = *(const v8h*)&xs_t[(s + 4) * XT + c0];
      const bool vL1 = (s >= 1), vL2 = (s >= 2), vR1 = (s <= 62), vR2 = (s <= 61);
#pragma unroll
      for (int r = 0; r < 8; ++r) {
        const float xm2 = (float)tp0[r];
        const float xm1 = (float)tp1[r];
        const float x0  = (float)tp2[r];
        const float xp1 = (float)tp3[r];
        const float xp2 = (float)tp4[r];
        const float sig = 1.f / (1.f + __expf(-accS[t][r]));
        if (b == 0) {
          const float avg = (xm1 + x0 + xp1) * (1.f / 3.f);   // zero-pad counted (PyTorch)
          float mx = x0;                                      // -inf pad: mask edges
          if (vL1) mx = fmaxf(mx, xm1);
          if (vR1) mx = fmaxf(mx, xp1);
          feat[t][r] = (avg + mx) * sig;
        } else {
          const float avg = (xm2 + xm1 + x0 + xp1 + xp2) * 0.2f;
          float mx = x0;
          if (vL1) mx = fmaxf(mx, xm1);
          if (vL2) mx = fmaxf(mx, xm2);
          if (vR1) mx = fmaxf(mx, xp1);
          if (vR2) mx = fmaxf(mx, xp2);
          feat[t][r] += (avg + mx) * sig;
        }
      }
    }
  }

  // =================== max over s, store out[p][n][c] ======================
  float* orow = out + (size_t)pn * C;
#pragma unroll
  for (int mt2 = 0; mt2 < 2; ++mt2) {
#pragma unroll
    for (int r = 0; r < 8; ++r) {
      float v = feat[mt2 * 4 + 0][r];
      v = fmaxf(v, feat[mt2 * 4 + 1][r]);
      v = fmaxf(v, feat[mt2 * 4 + 2][r]);
      v = fmaxf(v, feat[mt2 * 4 + 3][r]);
      v = fmaxf(v, __shfl_xor(v, 1, 32));        // reduce within 16-lane halves
      v = fmaxf(v, __shfl_xor(v, 2, 32));
      v = fmaxf(v, __shfl_xor(v, 4, 32));
      v = fmaxf(v, __shfl_xor(v, 8, 32));
      if ((lane & 15) == 0) {                    // lane0 -> c, lane16 -> c+8
        const int c = 32 * w + 16 * mt2 + r + ((lane & 16) ? 8 : 0);
        orow[c] = v;
      }
    }
  }
}

extern "C" void kernel_launch(void* const* d_in, const int* in_sizes, int n_in,
                              void* d_out, int out_size, void* d_ws, size_t ws_size,
                              hipStream_t stream) {
  (void)in_sizes; (void)n_in; (void)out_size; (void)d_ws; (void)ws_size;
  const float* x      = (const float*)d_in[0];
  const float* w3x1_a = (const float*)d_in[1];
  const float* w3x1_b = (const float*)d_in[2];
  const float* w3x3_a = (const float*)d_in[3];
  const float* w3x3_b = (const float*)d_in[4];
  float* out = (float*)d_out;

  // one block per (p, n): 16 * 128 = 2048 blocks, 256 threads (8 wave32s)
  tfa_fused_kernel<<<dim3(16 * 128), dim3(256), 0, stream>>>(
      x, w3x1_a, w3x1_b, w3x3_a, w3x3_b, out);
}
